// AliasFreeActivation_53042846105736
// MI455X (gfx1250) — compile-verified, matrix operand order checked by reading, not run on hardware
//
#include <hip/hip_runtime.h>

// ---- CDNA5 WMMA types -------------------------------------------------------
typedef __attribute__((ext_vector_type(16))) __bf16 v16bf;
typedef __attribute__((ext_vector_type(8)))  float  v8f;

// ---- problem constants ------------------------------------------------------
#define H_IN   36      // input H=W
#define UPL    144     // after x4 upsample
#define OUTD   52      // cropped output H=W
#define KX     64      // stage-1 K pad (36 -> 64, two K=32 steps)
#define NX     48      // X column pad (36 -> 48, three 16-tiles)
#define KZ     160     // stage-3 K pad (144 -> 160, five K=32 steps)
#define MD     64      // Dc row pad (52 -> 64, four 16-tiles)

#define U_ELEMS (UPL * KX)   // 9216  bf16 (18432 B)
#define D_ELEMS (MD  * KZ)   // 10240 bf16 (20480 B)

// ---- build dense banded operator matrices (bf16) in workspace ---------------
// U[o][i]  (144x36, padded to 144x64):  y_up[o] = sum_i U[o][i] x[i]
//   upfirdn up=4, pad=(13,10), 24-tap:  t = 13 + 4*i - o in [0,24) -> up[23-t]
// Dc[oc][j] (52x144, padded to 64x160): o = oc+10 (crop), down stride 2,
//   pad=(5,5), 12-tap: k = j - 2*oc - 15 in [0,12) -> down[11-k]
__global__ void build_mats_kernel(const float* __restrict__ up,
                                  const float* __restrict__ down,
                                  __bf16* __restrict__ Ug,
                                  __bf16* __restrict__ Dg) {
    int i = blockIdx.x * blockDim.x + threadIdx.x;
    if (i < U_ELEMS) {
        int o = i / KX, ii = i % KX;
        float v = 0.0f;
        if (ii < H_IN) {
            int t = 13 + 4 * ii - o;
            if (t >= 0 && t < 24) v = up[23 - t];
        }
        Ug[i] = (__bf16)v;
    }
    if (i < D_ELEMS) {
        int oc = i / KZ, j = i % KZ;
        float v = 0.0f;
        if (oc < OUTD && j < UPL) {
            int k = j - 2 * oc - 15;
            if (k >= 0 && k < 12) v = down[11 - k];
        }
        Dg[i] = (__bf16)v;
    }
}

// ---- CDNA5 async global->LDS copy (ASYNCcnt path, §15.18.3 op 98) ----------
// LDS destination address = low 32 bits of the flat pointer (aperture rule).
__device__ __forceinline__ void async_copy_b128(void* lds_ptr, const void* gptr) {
    unsigned lds_addr = (unsigned)(unsigned long long)lds_ptr;
    asm volatile("global_load_async_to_lds_b128 %0, %1, off"
                 :: "v"(lds_addr), "v"(gptr)
                 : "memory");
}
__device__ __forceinline__ void wait_async_zero() {
    asm volatile("s_wait_asynccnt 0x0" ::: "memory");
}

// ---- WMMA fragment helpers --------------------------------------------------
// 16-bit A-matrix 16x32 layout (ISA 7.12.2): lane r (0..15) holds row M=r,
// half=0 -> K 0..7 and 16..23 ; half=1 -> K 8..15 and 24..31.
// Two contiguous 16-byte chunks per lane -> two ds_load_b128.
// B (32x16) stored N-major (rows = N, cols = K) uses the identical pattern.
__device__ __forceinline__ v16bf load_frag(const __bf16* base, int row,
                                           int stride, int kbase) {
    union { v16bf v; uint4 q[2]; } u;
    const uint4* p = (const uint4*)(base + row * stride + kbase);
    u.q[0] = p[0];   // K .. K+7
    u.q[1] = p[2];   // K+16 .. K+23
    return u.v;
}

// One 16x16 output tile, K = 32*kSteps, A row-major [M][K], B N-major [N][K].
__device__ __forceinline__ v8f mm_tile(const __bf16* A, int as,
                                       const __bf16* B, int bs,
                                       int mT, int nT, int kSteps,
                                       int r, int half) {
    v8f acc = {};
    for (int ks = 0; ks < kSteps; ++ks) {
        int kbase = ks * 32 + half * 8;
        v16bf a = load_frag(A, mT * 16 + r, as, kbase);
        v16bf b = load_frag(B, nT * 16 + r, bs, kbase);
        acc = __builtin_amdgcn_wmma_f32_16x16x32_bf16(
                  false, a, false, b, (short)0, acc, false, false);
    }
    return acc;
}

// ---- fused alias-free activation: one workgroup per (b,c) image -------------
__global__ __launch_bounds__(256)
void afa_kernel(const float* __restrict__ x,
                const __bf16* __restrict__ Ug,
                const __bf16* __restrict__ Dg,
                float* __restrict__ out) {
    __shared__ __align__(16) __bf16 sU [U_ELEMS];   // U  [144][64]  (A of 1a, B of 1b)
    __shared__ __align__(16) __bf16 sD [D_ELEMS];   // Dc [64][160]  (A of 3a, B of 3b)
    __shared__ __align__(16) __bf16 sXt[NX  * KX];  // X^T padded [48][64]
    __shared__ __align__(16) __bf16 sT [UPL * KX];  // T = U*X    [144][64]
    __shared__ __align__(16) __bf16 sZt[UPL * KZ];  // lrelu(Y)^T [144][160]
    __shared__ __align__(16) __bf16 sW [MD  * KZ];  // W = Dc*Z   [64][160]

    const int tid  = threadIdx.x;
    const int lane = tid & 31;
    const int r    = lane & 15;
    const int half = lane >> 4;
    const int wv   = tid >> 5;              // 8 waves / workgroup

    const float* xb = x + (size_t)blockIdx.x * (H_IN * H_IN);

    // -- kick off async global->LDS copies of the constant operator matrices
    for (int c = tid; c < U_ELEMS / 8; c += 256)          // 1152 x 16B
        async_copy_b128(sU + c * 8, Ug + c * 8);
    for (int c = tid; c < D_ELEMS / 8; c += 256)          // 1280 x 16B
        async_copy_b128(sD + c * 8, Dg + c * 8);

    // -- meanwhile: load X transposed (with zero pad), zero scratch pads
    for (int i = tid; i < NX * KX; i += 256) {
        int n = i / KX, k = i % KX;
        float v = (n < H_IN && k < H_IN) ? xb[k * H_IN + n] : 0.0f;
        sXt[i] = (__bf16)v;
    }
    for (int i = tid; i < (UPL * KX) / 2; i += 256) ((unsigned*)sT )[i] = 0u;
    for (int i = tid; i < (UPL * KZ) / 2; i += 256) ((unsigned*)sZt)[i] = 0u;
    for (int i = tid; i < (MD  * KZ) / 2; i += 256) ((unsigned*)sW )[i] = 0u;

    wait_async_zero();
    __syncthreads();

    // -- stage 1a: T = U * Xp   (M=9 tiles, N=3 tiles, K=64)
    for (int job = wv; job < 9 * 3; job += 8) {
        int mT = job / 3, nT = job % 3;
        v8f acc = mm_tile(sU, KX, sXt, KX, mT, nT, 2, r, half);
        for (int j = 0; j < 8; ++j) {
            int m = mT * 16 + half * 8 + j;
            int n = nT * 16 + r;
            sT[m * KX + n] = (__bf16)acc[j];
        }
    }
    __syncthreads();

    // -- stage 1b: Y = T * U^T, fused leaky-ReLU, store transposed into sZt
    for (int job = wv; job < 9 * 9; job += 8) {
        int mT = job / 9, nT = job % 9;
        v8f acc = mm_tile(sT, KX, sU, KX, mT, nT, 2, r, half);
        for (int j = 0; j < 8; ++j) {
            int m = mT * 16 + half * 8 + j;
            int n = nT * 16 + r;
            float v = acc[j];
            v = (v < 0.0f) ? 0.2f * v : v;      // alias-free nonlinearity @144x144
            sZt[n * KZ + m] = (__bf16)v;
        }
    }
    __syncthreads();

    // -- stage 3a: W = Dc * Z   (M=4 tiles, N=9 tiles, K=160)
    for (int job = wv; job < 4 * 9; job += 8) {
        int mT = job / 9, nT = job % 9;
        v8f acc = mm_tile(sD, KZ, sZt, KZ, mT, nT, 5, r, half);
        for (int j = 0; j < 8; ++j) {
            int m = mT * 16 + half * 8 + j;
            int n = nT * 16 + r;
            sW[m * KZ + n] = (__bf16)acc[j];
        }
    }
    __syncthreads();

    // -- stage 3b: O = W * Dc^T (M=4, N=4, K=160), crop to 52x52, f32 out
    float* ob = out + (size_t)blockIdx.x * (OUTD * OUTD);
    for (int job = wv; job < 4 * 4; job += 8) {
        int mT = job / 4, nT = job % 4;
        v8f acc = mm_tile(sW, KZ, sD, KZ, mT, nT, 5, r, half);
        for (int j = 0; j < 8; ++j) {
            int m = mT * 16 + half * 8 + j;
            int n = nT * 16 + r;
            if (m < OUTD && n < OUTD) ob[m * OUTD + n] = acc[j];
        }
    }
}

// ---- entry ------------------------------------------------------------------
extern "C" void kernel_launch(void* const* d_in, const int* in_sizes, int n_in,
                              void* d_out, int out_size, void* d_ws, size_t ws_size,
                              hipStream_t stream) {
    const float* x    = (const float*)d_in[0];   // (16,512,36,36) f32
    const float* up   = (const float*)d_in[1];   // 24 taps
    const float* down = (const float*)d_in[2];   // 12 taps

    __bf16* Ug = (__bf16*)d_ws;                  // 144x64 bf16
    __bf16* Dg = Ug + U_ELEMS;                   // 64x160 bf16 (16B-aligned offset)

    build_mats_kernel<<<(D_ELEMS + 255) / 256, 256, 0, stream>>>(up, down, Ug, Dg);
    afa_kernel<<<16 * 512, 256, 0, stream>>>(x, Ug, Dg, (float*)d_out);
}
